// ResidualSyntaxGCN_31868657336591
// MI455X (gfx1250) — matrix operand order repaired and checked by reading, not compiled, not run on hardware
//
#include <hip/hip_runtime.h>
#include <hip/hip_bf16.h>

// ---------------------------------------------------------------------------
// ResidualSyntaxGCN forward for MI455X (gfx1250, wave32, WMMA).
// N=50000 nodes, E=800000 edges, G=256 graphs, H=64, L=4 residual blocks.
// Memory-bound: edge scatter dominates (~3.3 GB of gather + atomic traffic).
// GEMMs run on v_wmma_f32_16x16x32_f16; BN+ReLU is fused into the A-operand
// load (branch-free, templated), the GCN self-loop term is fused into the
// GEMM epilogue, and node buffers are row-padded to the GEMM grid so the
// whole GEMM kernel is branch-free (EXEC all-ones throughout).
// All float scatter-adds use hardware global_atomic_add_f32.
// ---------------------------------------------------------------------------

typedef __attribute__((ext_vector_type(16))) _Float16 v16h;
typedef __attribute__((ext_vector_type(8)))  float    v8f;

#define HID 64
#define NGRAPH 256

__device__ __forceinline__ void atomic_add_f32(float* p, float v) {
  unsafeAtomicAdd(p, v);   // -> global_atomic_add_f32 (no CAS loop)
}

// ----------------------------- utility kernels -----------------------------

__global__ void k_zero_f(float* __restrict__ p, int n) {
  int i = blockIdx.x * 256 + threadIdx.x;
  if (i < n) p[i] = 0.0f;
}

__global__ void k_edge_deg(const int* __restrict__ dst, int* __restrict__ deg, int E) {
  int e = blockIdx.x * 256 + threadIdx.x;
  if (e < E) atomicAdd(&deg[dst[e]], 1);
}

__global__ void k_dis(const int* __restrict__ deg, float* __restrict__ dis, int n) {
  int i = blockIdx.x * 256 + threadIdx.x;
  if (i < n) dis[i] = rsqrtf((float)deg[i] + 1.0f);
}

// h[i, c] = x[i, 0:3] @ proj_W + proj_b
__global__ void k_proj(const float* __restrict__ x, const float* __restrict__ W,
                       const float* __restrict__ b, float* __restrict__ h, int n) {
  int gid = blockIdx.x * 256 + threadIdx.x;
  if (gid >= n * HID) return;
  int i = gid >> 6, c = gid & 63;
  float v = b[c] + x[i*3+0]*W[0*HID+c] + x[i*3+1]*W[1*HID+c] + x[i*3+2]*W[2*HID+c];
  h[gid] = v;
}

// Pack 8 conv weight matrices (f32, [mat][k][n]) into f16 WMMA B-fragment
// order: Wp[((mat*2+s)*4+t)*32 + lane][j] ; k = 32s + (lane>=16)*8 + j + (j&8),
// n = t*16 + (lane&15).  One half per thread; 32768 threads total.
__global__ void k_pack_w(const float* __restrict__ W, _Float16* __restrict__ Wp) {
  int gid = blockIdx.x * 256 + threadIdx.x;
  if (gid >= 8 * 2 * 4 * 32 * 16) return;
  int j    = gid & 15;
  int lane = (gid >> 4) & 31;
  int t    = (gid >> 9) & 3;
  int s    = (gid >> 11) & 1;
  int m    = gid >> 12;
  int n    = t * 16 + (lane & 15);
  int k    = 32 * s + ((lane >> 4) * 8) + j + (j & 8);
  Wp[gid] = (_Float16)W[((size_t)m * HID + k) * HID + n];
}

// ------------------- WMMA conv GEMM + fused self-loop ----------------------
// 256 threads = 8 waves; wave w computes rows [blk*128 + w*16, +16), all 64
// output columns via 4 N-tiles (8 WMMAs / wave).
//   hw[row]  = act(A[row]) @ W
//   o[row]   = hw[row] * dis[row]^2 + bias       (aggregation buffer init)
// All node buffers are padded to the grid's row count, so there is no
// clamping or store predication: the kernel is completely branch-free.
// BN template variant applies per-column scale/shift + ReLU to A while
// converting f32 -> f16 (vectorized).

__device__ __forceinline__ float4 bn_act4(float4 a, float4 sc, float4 sh) {
  float4 r;
  r.x = fmaxf(fmaf(a.x, sc.x, sh.x), 0.0f);
  r.y = fmaxf(fmaf(a.y, sc.y, sh.y), 0.0f);
  r.z = fmaxf(fmaf(a.z, sc.z, sh.z), 0.0f);
  r.w = fmaxf(fmaf(a.w, sc.w, sh.w), 0.0f);
  return r;
}

template <bool BN>
__global__ void __launch_bounds__(256)
k_gemm_conv(const float* __restrict__ A, const _Float16* __restrict__ Wp,
            float* __restrict__ hw, float* __restrict__ o,
            const float* __restrict__ dis, const float* __restrict__ bias,
            const float* __restrict__ scale, const float* __restrict__ shift) {
  const int lane  = threadIdx.x & 31;
  const int wave  = threadIdx.x >> 5;
  const int rbase = (blockIdx.x * 8 + wave) * 16;
  const int mrow  = lane & 15;
  const int hi    = lane >> 4;
  const int arow  = rbase + mrow;

  const v16h* __restrict__ Bp = (const v16h*)Wp;
  v8f acc[4] = {};

#pragma unroll
  for (int s = 0; s < 2; ++s) {
    const int kb = 32 * s + hi * 8;         // multiple of 8 floats -> 32B aligned
    const float4* ap = (const float4*)(A + (size_t)arow * HID + kb);
    float4 a0 = ap[0], a1 = ap[1], a2 = ap[4], a3 = ap[5];
    if (BN) {
      const float4* scp = (const float4*)(scale + kb);
      const float4* shp = (const float4*)(shift + kb);
      a0 = bn_act4(a0, scp[0], shp[0]);
      a1 = bn_act4(a1, scp[1], shp[1]);
      a2 = bn_act4(a2, scp[4], shp[4]);
      a3 = bn_act4(a3, scp[5], shp[5]);
    }
    v16h afrag;
    afrag[0]  = (_Float16)a0.x; afrag[1]  = (_Float16)a0.y;
    afrag[2]  = (_Float16)a0.z; afrag[3]  = (_Float16)a0.w;
    afrag[4]  = (_Float16)a1.x; afrag[5]  = (_Float16)a1.y;
    afrag[6]  = (_Float16)a1.z; afrag[7]  = (_Float16)a1.w;
    afrag[8]  = (_Float16)a2.x; afrag[9]  = (_Float16)a2.y;
    afrag[10] = (_Float16)a2.z; afrag[11] = (_Float16)a2.w;
    afrag[12] = (_Float16)a3.x; afrag[13] = (_Float16)a3.y;
    afrag[14] = (_Float16)a3.z; afrag[15] = (_Float16)a3.w;
#pragma unroll
    for (int t = 0; t < 4; ++t) {
      v16h bfrag = Bp[(s * 4 + t) * 32 + lane];
      acc[t] = __builtin_amdgcn_wmma_f32_16x16x32_f16(
          false, afrag, false, bfrag, (short)0, acc[t], false, false);
    }
  }

  // epilogue: write hw and o = hw*dis^2 + bias (unconditional, padded rows)
  const int roff = hi * 8;
  const float4* dp = (const float4*)(dis + rbase + roff);   // 32B aligned
  const float4 dlo = dp[0], dhi = dp[1];
  float d2[8];
  d2[0] = dlo.x * dlo.x; d2[1] = dlo.y * dlo.y;
  d2[2] = dlo.z * dlo.z; d2[3] = dlo.w * dlo.w;
  d2[4] = dhi.x * dhi.x; d2[5] = dhi.y * dhi.y;
  d2[6] = dhi.z * dhi.z; d2[7] = dhi.w * dhi.w;
#pragma unroll
  for (int t = 0; t < 4; ++t) {
    const int col = t * 16 + mrow;
    const float bc = bias[col];
    float* hwp = hw + (size_t)(rbase + roff) * HID + col;
    float* op  = o  + (size_t)(rbase + roff) * HID + col;
#pragma unroll
    for (int r = 0; r < 8; ++r) {
      float v = acc[t][r];
      hwp[(size_t)r * HID] = v;
      op[(size_t)r * HID]  = v * d2[r] + bc;
    }
  }
}

// o[dst] += hw[src] * dis[src]*dis[dst]; 16 threads per edge, float4 each.
__global__ void k_edge_agg(const int* __restrict__ src, const int* __restrict__ dst,
                           const float* __restrict__ dis, const float* __restrict__ hw,
                           float* __restrict__ o, int E) {
  int gid = blockIdx.x * 256 + threadIdx.x;
  if (gid >= E * 16) return;
  int e = gid >> 4, q = gid & 15;
  int s = src[e], d = dst[e];
  float coef = dis[s] * dis[d];
  const float4 v = *(const float4*)(hw + (size_t)s * HID + q * 4);
  float* op = o + (size_t)d * HID + q * 4;
  atomic_add_f32(op + 0, v.x * coef);
  atomic_add_f32(op + 1, v.y * coef);
  atomic_add_f32(op + 2, v.z * coef);
  atomic_add_f32(op + 3, v.w * coef);
}

// Per-channel sum / sum-of-squares over N rows (64 channels).
// Block handles 128 rows; LDS tree then one atomicAdd pair per channel.
__global__ void k_bn_stats(const float* __restrict__ x, int n, float* __restrict__ sums) {
  __shared__ float ls[256], lq[256];
  const int c = threadIdx.x & 63, rg = threadIdx.x >> 6;
  const int base = blockIdx.x * 128;
  int end = base + 128; if (end > n) end = n;
  float s = 0.0f, q = 0.0f;
  for (int r = base + rg; r < end; r += 4) {
    float v = x[(size_t)r * HID + c];
    s += v; q += v * v;
  }
  ls[threadIdx.x] = s; lq[threadIdx.x] = q;
  __syncthreads();
  if (threadIdx.x < 64) {
    s = ls[c] + ls[c + 64] + ls[c + 128] + ls[c + 192];
    q = lq[c] + lq[c + 64] + lq[c + 128] + lq[c + 192];
    atomic_add_f32(&sums[c], s);
    atomic_add_f32(&sums[64 + c], q);
  }
}

// scale = g * rsqrt(var + eps); shift = b - mean * scale   (biased variance)
__global__ void k_bn_finalize(const float* __restrict__ sums, float invn,
                              const float* __restrict__ g, const float* __restrict__ b,
                              float* __restrict__ scale, float* __restrict__ shift) {
  int c = threadIdx.x;
  float mean = sums[c] * invn;
  float var  = sums[64 + c] * invn - mean * mean;
  float sc   = g[c] * rsqrtf(var + 1e-5f);
  scale[c] = sc;
  shift[c] = b[c] - mean * sc;
}

// h = relu(bn2(o) + h)   (residual, in place)
__global__ void k_residual(const float* __restrict__ o, const float* __restrict__ scale,
                           const float* __restrict__ shift, float* __restrict__ h, int n) {
  int gid = blockIdx.x * 256 + threadIdx.x;
  if (gid >= n * HID) return;
  int c = gid & 63;
  h[gid] = fmaxf(fmaf(o[gid], scale[c], shift[c]) + h[gid], 0.0f);
}

// Segment mean/max pooling scatter.  h >= 0 (post-ReLU) so integer atomicMax
// on raw float bits is exact, and a 0-initialized max matches the reference's
// empty-graph guard.
__global__ void k_pool(const float* __restrict__ h, const int* __restrict__ batch,
                       float* __restrict__ psum, float* __restrict__ pmax,
                       float* __restrict__ pcnt, int n) {
  int gid = blockIdx.x * 256 + threadIdx.x;
  if (gid >= n * 16) return;
  int i = gid >> 4, q = gid & 15;
  int g = batch[i];
  const float4 v = *(const float4*)(h + (size_t)i * HID + q * 4);
  float* sp = psum + (size_t)g * HID + q * 4;
  atomic_add_f32(sp + 0, v.x); atomic_add_f32(sp + 1, v.y);
  atomic_add_f32(sp + 2, v.z); atomic_add_f32(sp + 3, v.w);
  int* mp = (int*)(pmax + (size_t)g * HID + q * 4);
  atomicMax(mp + 0, __float_as_int(v.x));
  atomicMax(mp + 1, __float_as_int(v.y));
  atomicMax(mp + 2, __float_as_int(v.z));
  atomicMax(mp + 3, __float_as_int(v.w));
  if (q == 0) atomic_add_f32(&pcnt[g], 1.0f);
}

// z[g] = concat(mean, max)
__global__ void k_poolfin(const float* __restrict__ psum, const float* __restrict__ pmax,
                          const float* __restrict__ pcnt, float* __restrict__ z) {
  int gid = blockIdx.x * 256 + threadIdx.x;
  if (gid >= NGRAPH * HID) return;
  int g = gid >> 6, c = gid & 63;
  float cnt = fmaxf(pcnt[g], 1.0f);
  z[(size_t)g * 128 + c]      = psum[gid] / cnt;
  z[(size_t)g * 128 + 64 + c] = pmax[gid];
}

// ----------------------------- tiny MLP head -------------------------------

__global__ void k_head_gemm(const float* __restrict__ A, const float* __restrict__ W,
                            const float* __restrict__ b, float* __restrict__ out,
                            int rows, int K, int Cout) {
  int gid = blockIdx.x * 256 + threadIdx.x;
  if (gid >= rows * Cout) return;
  int i = gid / Cout, j = gid % Cout;
  float acc = b[j];
  for (int k = 0; k < K; ++k) acc += A[(size_t)i * K + k] * W[(size_t)k * Cout + j];
  out[gid] = acc;
}

__global__ void k_bn_small(const float* __restrict__ x, int rows, int C,
                           const float* __restrict__ g, const float* __restrict__ b,
                           float* __restrict__ scale, float* __restrict__ shift) {
  int c = threadIdx.x;
  if (c >= C) return;
  float s = 0.0f, q = 0.0f;
  for (int r = 0; r < rows; ++r) { float v = x[(size_t)r * C + c]; s += v; q += v * v; }
  float mean = s / rows;
  float var  = q / rows - mean * mean;
  float sc   = g[c] * rsqrtf(var + 1e-5f);
  scale[c] = sc;
  shift[c] = b[c] - mean * sc;
}

__global__ void k_apply_small(float* __restrict__ x, const float* __restrict__ scale,
                              const float* __restrict__ shift, int n, int C) {
  int gid = blockIdx.x * 256 + threadIdx.x;
  if (gid >= n) return;
  int c = gid % C;
  x[gid] = fmaxf(fmaf(x[gid], scale[c], shift[c]), 0.0f);
}

// ----------------------------- launch --------------------------------------

static inline unsigned gdiv(long n, int b) { return (unsigned)((n + b - 1) / b); }

extern "C" void kernel_launch(void* const* d_in, const int* in_sizes, int n_in,
                              void* d_out, int out_size, void* d_ws, size_t ws_size,
                              hipStream_t stream) {
  const float* x      = (const float*)d_in[0];
  const int*   edge   = (const int*)d_in[1];      // [2, E] flat: src | dst
  const int*   batch  = (const int*)d_in[2];
  const float* proj_W = (const float*)d_in[3];
  const float* proj_b = (const float*)d_in[4];
  const float* conv_W = (const float*)d_in[5];    // [4][2][64][64]
  const float* conv_b = (const float*)d_in[6];    // [4][2][64]
  const float* bn_g   = (const float*)d_in[7];
  const float* bn_b   = (const float*)d_in[8];
  const float* lin1_W = (const float*)d_in[9];
  const float* lin1_b = (const float*)d_in[10];
  const float* lin2_W = (const float*)d_in[11];
  const float* lin2_b = (const float*)d_in[12];
  const float* lin3_W = (const float*)d_in[13];
  const float* lin3_b = (const float*)d_in[14];
  const float* bnf1_g = (const float*)d_in[15];
  const float* bnf1_b = (const float*)d_in[16];
  const float* bnf2_g = (const float*)d_in[17];
  const float* bnf2_b = (const float*)d_in[18];

  const int N     = in_sizes[0] / 3;
  const int E     = in_sizes[1] / 2;
  const int Npad  = ((N + 127) / 128) * 128;      // GEMM grid row coverage
  const int* src  = edge;
  const int* dst  = edge + E;

  // ---- workspace carve (256B aligned) ----
  char* wp = (char*)d_ws;
  auto carve = [&](size_t bytes) -> void* {
    void* p = (void*)wp;
    wp += (bytes + 255) & ~(size_t)255;
    return p;
  };
  float*     h     = (float*)carve((size_t)Npad * HID * 4);
  float*     hwb   = (float*)carve((size_t)Npad * HID * 4);
  float*     o     = (float*)carve((size_t)Npad * HID * 4);
  float*     dis   = (float*)carve((size_t)Npad * 4);
  int*       degi  = (int*)  carve((size_t)N * 4);
  _Float16*  Wp    = (_Float16*)carve((size_t)8 * 2 * 4 * 32 * 16 * 2);
  float*     stats = (float*)carve(128 * 4);
  float*     sc1   = (float*)carve(64 * 4);
  float*     sh1   = (float*)carve(64 * 4);
  float*     sc2   = (float*)carve(64 * 4);
  float*     sh2   = (float*)carve(64 * 4);
  float*     psum  = (float*)carve((size_t)NGRAPH * HID * 4);
  float*     pmax  = (float*)carve((size_t)NGRAPH * HID * 4);
  float*     pcnt  = (float*)carve((size_t)NGRAPH * 4);
  float*     z     = (float*)carve((size_t)NGRAPH * 128 * 4);
  float*     z1    = (float*)carve((size_t)NGRAPH * 64 * 4);
  float*     z2    = (float*)carve((size_t)NGRAPH * 32 * 4);
  (void)ws_size; (void)n_in; (void)out_size;

  const unsigned gNH   = gdiv((long)N * HID, 256);
  const unsigned gE16  = gdiv((long)E * 16, 256);
  const unsigned gGemm = (unsigned)(Npad / 128);        // 8 waves x 16 rows / block
  const unsigned gStat = gdiv(N, 128);
  const float    invN  = 1.0f / (float)N;

  // ---- degrees & symmetric norm coefficients ----
  k_zero_f  <<<gdiv(Npad, 256), 256, 0, stream>>>(dis, Npad);   // incl. pad rows
  k_zero_f  <<<gdiv(N, 256), 256, 0, stream>>>((float*)degi, N);
  k_edge_deg<<<gdiv(E, 256), 256, 0, stream>>>(dst, degi, E);
  k_dis     <<<gdiv(N, 256), 256, 0, stream>>>(degi, dis, N);

  // ---- input projection & weight packing ----
  k_proj  <<<gNH, 256, 0, stream>>>(x, proj_W, proj_b, h, N);
  k_pack_w<<<gdiv(8 * 2 * 4 * 32 * 16, 256), 256, 0, stream>>>(conv_W, Wp);

  // ---- 4 residual GCN blocks ----
  for (int l = 0; l < 4; ++l) {
    const _Float16* W1 = Wp + (size_t)(l * 2 + 0) * (2 * 4 * 32 * 16);
    const _Float16* W2 = Wp + (size_t)(l * 2 + 1) * (2 * 4 * 32 * 16);

    // conv1: hw = h @ W1, o = hw*dis^2 + b (fused); then o += edge messages
    k_gemm_conv<false><<<gGemm, 256, 0, stream>>>(
        h, W1, hwb, o, dis, conv_b + (l * 2 + 0) * HID, nullptr, nullptr);
    k_edge_agg<<<gE16, 256, 0, stream>>>(src, dst, dis, hwb, o, E);

    // BN1 stats -> (scale, shift); application fused into next GEMM's A read
    k_zero_f     <<<1, 128, 0, stream>>>(stats, 128);
    k_bn_stats   <<<gStat, 256, 0, stream>>>(o, N, stats);
    k_bn_finalize<<<1, 64, 0, stream>>>(stats, invN,
                                        bn_g + (l * 2 + 0) * HID, bn_b + (l * 2 + 0) * HID,
                                        sc1, sh1);

    // conv2: hw = relu(bn1(o)) @ W2, o = hw*dis^2 + b (fused); o += messages
    k_gemm_conv<true><<<gGemm, 256, 0, stream>>>(
        o, W2, hwb, o, dis, conv_b + (l * 2 + 1) * HID, sc1, sh1);
    k_edge_agg<<<gE16, 256, 0, stream>>>(src, dst, dis, hwb, o, E);

    // BN2 stats, then residual: h = relu(bn2(o) + h)
    k_zero_f     <<<1, 128, 0, stream>>>(stats, 128);
    k_bn_stats   <<<gStat, 256, 0, stream>>>(o, N, stats);
    k_bn_finalize<<<1, 64, 0, stream>>>(stats, invN,
                                        bn_g + (l * 2 + 1) * HID, bn_b + (l * 2 + 1) * HID,
                                        sc2, sh2);
    k_residual<<<gNH, 256, 0, stream>>>(o, sc2, sh2, h, N);
  }

  // ---- segment mean/max pooling ----
  k_zero_f <<<gdiv(NGRAPH * HID, 256), 256, 0, stream>>>(psum, NGRAPH * HID);
  k_zero_f <<<gdiv(NGRAPH * HID, 256), 256, 0, stream>>>(pmax, NGRAPH * HID);
  k_zero_f <<<1, 256, 0, stream>>>(pcnt, NGRAPH);
  k_pool   <<<gdiv((long)N * 16, 256), 256, 0, stream>>>(h, batch, psum, pmax, pcnt, N);
  k_poolfin<<<gdiv(NGRAPH * HID, 256), 256, 0, stream>>>(psum, pmax, pcnt, z);

  // ---- MLP head (tiny; 256 rows) ----
  k_head_gemm  <<<gdiv(NGRAPH * 64, 256), 256, 0, stream>>>(z, lin1_W, lin1_b, z1, NGRAPH, 128, 64);
  k_bn_small   <<<1, 64, 0, stream>>>(z1, NGRAPH, 64, bnf1_g, bnf1_b, sc1, sh1);
  k_apply_small<<<gdiv(NGRAPH * 64, 256), 256, 0, stream>>>(z1, sc1, sh1, NGRAPH * 64, 64);

  k_head_gemm  <<<gdiv(NGRAPH * 32, 256), 256, 0, stream>>>(z1, lin2_W, lin2_b, z2, NGRAPH, 64, 32);
  k_bn_small   <<<1, 32, 0, stream>>>(z2, NGRAPH, 32, bnf2_g, bnf2_b, sc2, sh2);
  k_apply_small<<<gdiv(NGRAPH * 32, 256), 256, 0, stream>>>(z2, sc2, sh2, NGRAPH * 32, 32);

  k_head_gemm<<<gdiv(NGRAPH * 2, 256), 256, 0, stream>>>(z2, lin3_W, lin3_b, (float*)d_out,
                                                         NGRAPH, 32, 2);
}
